// Memory_14611478741075
// MI455X (gfx1250) — matrix operand order repaired and checked by reading, not compiled, not run on hardware
//
#include <hip/hip_runtime.h>
#include <hip/hip_bf16.h>

// ---------------------------------------------------------------------------
// VQ-VAE keypoint tokenizer for MI455X (gfx1250).
// GEMMs (512x512 channel MLPs, feat_embed, dec_start) run on v_wmma_f32_16x16x32_f16
// with fp32 accumulate. B (weight) tiles are staged per-block into LDS with
// double-buffered GLOBAL_LOAD_ASYNC_TO_LDS_B128 (ASYNCcnt pipeline) to remove
// the 4x per-wave redundancy; A tiles are loaded f32 and converted in-register.
// Small ops (layernorm, token MLPs, VQ, EMA) are wave32 VALU kernels.
//
// Input flattening assumption (JAX pytree: dict keys sorted alphabetically):
//   0 codebook[34,16,512]  1 ema_cluster_size[34,16]  2 ema_w[34,16,512]
//   3.. params: dec_layers[0]{ch1.b,ch1.w,ch2.b,ch2.w,ln1.b,ln1.g,ln2.b,ln2.g,
//                             tok1.b,tok1.w,tok2.b,tok2.w},
//       dec_ln{b,g}, dec_start{b,w}, dec_token_mlp{b,w},
//       enc_layers[0..3]{same 12}, enc_ln{b,g}, enc_start{b,w},
//       feat_embed{b,w}, recover{b,w}, token_mlp{b,w}
//   79 x[4096,17,2]
// Output: out[4096*17*2] | vq_loss[1] | codebook_new[278528] | cs[544] | ema_w_new[278528]
// ---------------------------------------------------------------------------

typedef __attribute__((ext_vector_type(16))) _Float16 v16h;
typedef __attribute__((ext_vector_type(8)))  _Float16 v8h;
typedef __attribute__((ext_vector_type(8)))  float    v8f;
typedef __attribute__((ext_vector_type(4)))  float    v4f;

#define NB   4096
#define NP   17
#define NT34 34
#define NK   16
#define ND   512
#define RP   (NB*NP)     // 69632
#define RT   (NB*NT34)   // 139264

__device__ __forceinline__ float gelu_exact(float x) {
    return 0.5f * x * (1.0f + erff(x * 0.70710678118654752440f));
}

// ------------------------- f32 -> f16 weight convert ------------------------
__global__ __launch_bounds__(256) void k_f32_to_f16(const float* __restrict__ s,
                                                    _Float16* __restrict__ d, int n) {
    int i = blockIdx.x * 256 + threadIdx.x;
    if (i < n) d[i] = (_Float16)s[i];
}

// ------------------------------- WMMA GEMM ---------------------------------
// C[M,N] = act( A[M,K](f32) @ W[N,K]^T(f16) + bias ) (+ R[M,N])
// block = 128 threads = 4 waves; wave -> 16 rows x (16*NTI) cols.
// B tile (16*NTI cols x 32 K of f16) staged in LDS via async-to-LDS DMA,
// double-buffered across the K loop. grid = (M/64, N/(16*NTI)).
// M%64==0, N%(16*NTI)==0, K%32==0.
template<int NTI, bool GELU, bool RES>
__global__ __launch_bounds__(128)
void k_gemm_wmma(const float* __restrict__ A, const _Float16* __restrict__ W,
                 const float* __restrict__ bias, const float* __restrict__ R,
                 float* __restrict__ C, int M, int N, int K) {
    constexpr int NL = 16 * NTI;          // local cols per block
    constexpr int ROWH = 40;              // padded row stride in halfs (80B)
    constexpr int CHUNKS = NL * 4;        // 16B chunks per stage (4 per row)
    constexpr int PERLANE = CHUNKS / 128; // async issues per lane per stage

    __shared__ _Float16 smB[2][NL * ROWH];

    const int lane = threadIdx.x & 31;
    const int wave = threadIdx.x >> 5;
    const int hf   = lane >> 4;       // wave half (0/1)
    const int l16  = lane & 15;
    const int m0   = blockIdx.x * 64 + wave * 16;
    const int n0   = blockIdx.y * NL;
    const int mRow = m0 + l16;        // A row this lane loads

    // async-stage the 16*NTI x 32 f16 weight tile for k0 into smB[buf]
    auto issueB = [&](int k0, int buf) {
#pragma unroll
        for (int q = 0; q < PERLANE; ++q) {
            const int c    = threadIdx.x + q * 128;   // chunk id
            const int n    = c >> 2;                  // local col
            const int part = c & 3;                   // 8-half group within row
            unsigned lds = (unsigned)(uintptr_t)&smB[buf][n * ROWH + part * 8];
            unsigned long long g =
                (unsigned long long)(uintptr_t)(W + (size_t)(n0 + n) * K + k0 + part * 8);
            asm volatile("global_load_async_to_lds_b128 %0, %1, off"
                         :: "v"(lds), "v"(g) : "memory");
        }
    };

    v8f acc[NTI] = {};
    const int S = K >> 5;             // K/32 stages
    issueB(0, 0);

    for (int s = 0; s < S; ++s) {
        const int k0 = s << 5;
        if (s + 1 < S) {
            issueB(k0 + 32, (s + 1) & 1);
            asm volatile("s_wait_asynccnt %0" :: "n"(PERLANE) : "memory");
        } else {
            asm volatile("s_wait_asynccnt 0x0" ::: "memory");
        }
        __syncthreads();              // stage s resident & visible block-wide

        // ---- A fragment: lane holds row mRow, K = {k0+8h..+7, k0+16+8h..+7}
        const float* ap = A + (size_t)mRow * K + k0 + 8 * hf;
        v4f a0 = *(const v4f*)(ap);
        v4f a1 = *(const v4f*)(ap + 4);
        v4f a2 = *(const v4f*)(ap + 16);
        v4f a3 = *(const v4f*)(ap + 20);
        v16h af;
#pragma unroll
        for (int i = 0; i < 4; ++i) {
            af[i]      = (_Float16)a0[i];
            af[4 + i]  = (_Float16)a1[i];
            af[8 + i]  = (_Float16)a2[i];
            af[12 + i] = (_Float16)a3[i];
        }
        if (k0 + 32 < K)  // next A tile into cache: global_prefetch_b8
            __builtin_prefetch((const void*)(A + (size_t)mRow * K + k0 + 32), 0, 1);

        const _Float16* sb = &smB[s & 1][0];
#pragma unroll
        for (int t = 0; t < NTI; ++t) {
            // ---- B fragment from LDS: lane holds col l16+16t, K = {8h..8h+7, 16+8h..16+8h+7}
            const _Float16* bp = sb + (l16 + 16 * t) * ROWH + 8 * hf;
            v8h blo = *(const v8h*)(bp);
            v8h bhi = *(const v8h*)(bp + 16);
            v16h bf;
#pragma unroll
            for (int i = 0; i < 8; ++i) { bf[i] = blo[i]; bf[8 + i] = bhi[i]; }
            acc[t] = __builtin_amdgcn_wmma_f32_16x16x32_f16(
                false, af, false, bf, (short)0, acc[t], false, false);
        }
        __syncthreads();              // protect buffer (s&1) before re-fill at s+2
    }

#pragma unroll
    for (int t = 0; t < NTI; ++t) {
        const int n = n0 + l16 + 16 * t;
        const float bv = bias[n];
#pragma unroll
        for (int v = 0; v < 8; ++v) {
            const int m = m0 + 8 * hf + v;    // C/D layout: m = vgpr + 8*half
            float c = acc[t][v] + bv;
            if (GELU) c = gelu_exact(c);
            if (RES)  c += R[(size_t)m * N + n];
            C[(size_t)m * N + n] = c;
        }
    }
}

// ------------------------------ enc_start ----------------------------------
__global__ __launch_bounds__(256)
void k_enc_start(const float* __restrict__ x, const float* __restrict__ w,
                 const float* __restrict__ b, float* __restrict__ f) {
    const int row = blockIdx.x;                    // B*17
    const float x0 = x[row * 2], x1 = x[row * 2 + 1];
    for (int o = threadIdx.x; o < ND; o += 256)
        f[(size_t)row * ND + o] = w[o * 2] * x0 + w[o * 2 + 1] * x1 + b[o];
}

// ------------------------------ layernorms ---------------------------------
__global__ __launch_bounds__(256)
void k_ln512(const float* __restrict__ X, const float* __restrict__ g,
             const float* __restrict__ bb, float* __restrict__ Y) {
    const int row = blockIdx.x;
    const float* x = X + (size_t)row * ND;
    float v0 = x[threadIdx.x], v1 = x[threadIdx.x + 256];
    float s = v0 + v1, s2 = v0 * v0 + v1 * v1;
    for (int o = 16; o > 0; o >>= 1) { s += __shfl_down(s, o, 32); s2 += __shfl_down(s2, o, 32); }
    __shared__ float sh[16];
    const int w = threadIdx.x >> 5, l = threadIdx.x & 31;
    if (l == 0) { sh[w] = s; sh[8 + w] = s2; }
    __syncthreads();
    if (threadIdx.x == 0) {
        float ts = 0, ts2 = 0;
        for (int i = 0; i < 8; ++i) { ts += sh[i]; ts2 += sh[8 + i]; }
        sh[0] = ts / 512.f; sh[1] = ts2 / 512.f;
    }
    __syncthreads();
    const float mean = sh[0];
    const float inv  = rsqrtf(sh[1] - mean * mean + 1e-5f);
    Y[(size_t)row * ND + threadIdx.x]       = (v0 - mean) * inv * g[threadIdx.x] + bb[threadIdx.x];
    Y[(size_t)row * ND + threadIdx.x + 256] = (v1 - mean) * inv * g[threadIdx.x + 256] + bb[threadIdx.x + 256];
}

__global__ __launch_bounds__(256)
void k_ln32(const float* __restrict__ X, const float* __restrict__ g,
            const float* __restrict__ bb, float* __restrict__ Y, int rows) {
    const int w = threadIdx.x >> 5, l = threadIdx.x & 31;
    const int row = blockIdx.x * 8 + w;
    if (row >= rows) return;
    float v = X[(size_t)row * 32 + l];
    float s = v, s2 = v * v;
    for (int o = 16; o > 0; o >>= 1) { s += __shfl_down(s, o, 32); s2 += __shfl_down(s2, o, 32); }
    s = __shfl(s, 0, 32); s2 = __shfl(s2, 0, 32);
    const float mean = s / 32.f;
    Y[(size_t)row * 32 + l] = (v - mean) * rsqrtf(s2 / 32.f - mean * mean + 1e-5f) * g[l] + bb[l];
}

// ------------------- token-mixing MLP (17 -> gelu(64) -> 17) ---------------
template<int HD>
__global__ __launch_bounds__(256)
void k_tokenmix(const float* __restrict__ Yln, const float* __restrict__ Xin,
                const float* __restrict__ w1, const float* __restrict__ b1,
                const float* __restrict__ w2, const float* __restrict__ b2,
                float* __restrict__ Rout, int Btot) {
    __shared__ float sw1[64 * 17], sw2[17 * 64], sb1[64], sb2[17];
    for (int i = threadIdx.x; i < 64 * 17; i += 256) { sw1[i] = w1[i]; sw2[i] = w2[i]; }
    for (int i = threadIdx.x; i < 64; i += 256) sb1[i] = b1[i];
    for (int i = threadIdx.x; i < 17; i += 256) sb2[i] = b2[i];
    __syncthreads();
    const int idx = blockIdx.x * 256 + threadIdx.x;   // over B*HD
    if (idx >= Btot * HD) return;
    const int b = idx / HD, h = idx % HD;
    const float* yb = Yln + (size_t)b * NP * HD + h;
    float t[NP];
#pragma unroll
    for (int p = 0; p < NP; ++p) t[p] = yb[p * HD];
    float u[64];
#pragma unroll
    for (int j = 0; j < 64; ++j) {
        float a = sb1[j];
#pragma unroll
        for (int p = 0; p < NP; ++p) a += sw1[j * 17 + p] * t[p];
        u[j] = gelu_exact(a);
    }
    const float* xb = Xin + (size_t)b * NP * HD + h;
    float* rb = Rout + (size_t)b * NP * HD + h;
#pragma unroll
    for (int p = 0; p < NP; ++p) {
        float a = sb2[p];
#pragma unroll
        for (int j = 0; j < 64; ++j) a += sw2[p * 64 + j] * u[j];
        rb[p * HD] = xb[p * HD] + a;
    }
}

// ------------- token projection across parts dim (TIN -> TOUT) -------------
template<int TIN, int TOUT>
__global__ __launch_bounds__(256)
void k_tokproj(const float* __restrict__ In, const float* __restrict__ w,
               const float* __restrict__ bias, float* __restrict__ Out, int Btot) {
    __shared__ float sw[TOUT * TIN], sb[TOUT];
    for (int i = threadIdx.x; i < TOUT * TIN; i += 256) sw[i] = w[i];
    for (int i = threadIdx.x; i < TOUT; i += 256) sb[i] = bias[i];
    __syncthreads();
    const int idx = blockIdx.x * 256 + threadIdx.x;   // over B*512
    if (idx >= Btot * ND) return;
    const int b = idx / ND, h = idx % ND;
    float t[TIN];
    const float* ib = In + (size_t)b * TIN * ND + h;
#pragma unroll
    for (int p = 0; p < TIN; ++p) t[p] = ib[p * ND];
    float* ob = Out + (size_t)b * TOUT * ND + h;
#pragma unroll
    for (int q = 0; q < TOUT; ++q) {
        float a = sb[q];
#pragma unroll
        for (int p = 0; p < TIN; ++p) a += sw[q * TIN + p] * t[p];
        ob[q * ND] = a;
    }
}

// ------------------------------- VQ stage ----------------------------------
__global__ __launch_bounds__(256)
void k_cnorm(const float* __restrict__ cb, float* __restrict__ cnorm) {  // 544 rows
    const int row = blockIdx.x;
    const float* c = cb + (size_t)row * ND;
    float v0 = c[threadIdx.x], v1 = c[threadIdx.x + 256];
    float s = v0 * v0 + v1 * v1;
    for (int o = 16; o > 0; o >>= 1) s += __shfl_down(s, o, 32);
    __shared__ float sh[8];
    if ((threadIdx.x & 31) == 0) sh[threadIdx.x >> 5] = s;
    __syncthreads();
    if (threadIdx.x == 0) {
        float t2 = 0;
        for (int i = 0; i < 8; ++i) t2 += sh[i];
        cnorm[row] = t2;
    }
}

__global__ __launch_bounds__(256)
void k_vq_argmin(const float* __restrict__ ef, const float* __restrict__ cb,
                 const float* __restrict__ cnorm, int* __restrict__ idxO,
                 float* __restrict__ counts) {
    const int w = threadIdx.x >> 5, l = threadIdx.x & 31;
    const int row = blockIdx.x * 8 + w;               // b*34 + t
    if (row >= RT) return;
    const int t = row % NT34;
    const float* e = ef + (size_t)row * ND;
    float best = 3.4e38f; int bi = 0;
    for (int k = 0; k < NK; ++k) {
        const float* c = cb + ((size_t)t * NK + k) * ND;
        float s = 0;
        for (int d = l; d < ND; d += 32) s += e[d] * c[d];
        for (int o = 16; o > 0; o >>= 1) s += __shfl_down(s, o, 32);
        s = __shfl(s, 0, 32);
        const float dist = cnorm[t * NK + k] - 2.0f * s;  // ||e||^2 const in argmin
        if (dist < best) { best = dist; bi = k; }
    }
    if (l == 0) { idxO[row] = bi; atomicAdd(&counts[t * NK + bi], 1.0f); }
}

__global__ __launch_bounds__(128)
void k_vq_scatter(const float* __restrict__ ef, const float* __restrict__ cb,
                  const int* __restrict__ idxO, float* __restrict__ tf,
                  float* __restrict__ dw, float* __restrict__ lossAcc) {
    const int row = blockIdx.x;                        // b*34 + t
    const int t = row % NT34;
    const int k = idxO[row];
    const float* e = ef + (size_t)row * ND;
    const float* c = cb + ((size_t)t * NK + k) * ND;
    float* dwp = dw + ((size_t)t * NK + k) * ND;
    float* q = tf + (size_t)row * ND;
    float ls = 0;
    for (int d = threadIdx.x; d < ND; d += 128) {
        const float ev = e[d], cv = c[d];
        q[d] = cv;                                     // straight-through: tf == q
        atomicAdd(&dwp[d], ev);
        const float df = cv - ev;
        ls += df * df;
    }
    for (int o = 16; o > 0; o >>= 1) ls += __shfl_down(ls, o, 32);
    __shared__ float sh[4];
    if ((threadIdx.x & 31) == 0) sh[threadIdx.x >> 5] = ls;
    __syncthreads();
    if (threadIdx.x == 0) atomicAdd(lossAcc, sh[0] + sh[1] + sh[2] + sh[3]);
}

__global__ void k_cs(const float* __restrict__ ema_cs, const float* __restrict__ counts,
                     float* __restrict__ csOut) {     // grid 34, block 16
    __shared__ float s0[NK];
    const int t = blockIdx.x, k = threadIdx.x;
    const float cs0 = ema_cs[t * NK + k] * 0.9f + 0.1f * counts[t * NK + k];
    s0[k] = cs0;
    __syncthreads();
    float n = 0;
    for (int i = 0; i < NK; ++i) n += s0[i];
    csOut[t * NK + k] = (cs0 + 1e-5f) / (n + NK * 1e-5f) * n;
}

__global__ void k_loss_fin(const float* __restrict__ acc, float* __restrict__ out) {
    out[0] = acc[0] / ((float)RT * (float)ND);
}

__global__ __launch_bounds__(256)
void k_ema(const float* __restrict__ ema_w, const float* __restrict__ dw,
           const float* __restrict__ cs, float* __restrict__ ema_w_new,
           float* __restrict__ cb_new) {              // grid 544
    const int row = blockIdx.x;
    const float c = cs[row];
    const size_t base = (size_t)row * ND;
    for (int d = threadIdx.x; d < ND; d += 256) {
        const float v = ema_w[base + d] * 0.9f + 0.1f * dw[base + d];
        ema_w_new[base + d] = v;
        cb_new[base + d] = v / c;
    }
}

// ------------------ decoder channel MLP (32 -> gelu(64) -> 32) -------------
__global__ __launch_bounds__(256)
void k_dec_chmlp(const float* __restrict__ Z, const float* __restrict__ Rin,
                 const float* __restrict__ w1, const float* __restrict__ b1,
                 const float* __restrict__ w2, const float* __restrict__ b2,
                 float* __restrict__ Out, int rows) {
    __shared__ float s1[64 * 32], s2[32 * 64], sb1[64], sb2[32];
    for (int i = threadIdx.x; i < 64 * 32; i += 256) { s1[i] = w1[i]; s2[i] = w2[i]; }
    if (threadIdx.x < 64) sb1[threadIdx.x] = b1[threadIdx.x];
    if (threadIdx.x < 32) sb2[threadIdx.x] = b2[threadIdx.x];
    __syncthreads();
    const int r = blockIdx.x * 256 + threadIdx.x;
    if (r >= rows) return;
    const float* z = Z + (size_t)r * 32;
    float zz[32];
#pragma unroll
    for (int i = 0; i < 32; ++i) zz[i] = z[i];
    float u[64];
#pragma unroll
    for (int j = 0; j < 64; ++j) {
        float a = sb1[j];
#pragma unroll
        for (int i = 0; i < 32; ++i) a += s1[j * 32 + i] * zz[i];
        u[j] = gelu_exact(a);
    }
    const float* rr = Rin + (size_t)r * 32;
    float* o = Out + (size_t)r * 32;
#pragma unroll
    for (int i = 0; i < 32; ++i) {
        float a = sb2[i];
#pragma unroll
        for (int j = 0; j < 64; ++j) a += s2[i * 64 + j] * u[j];
        o[i] = rr[i] + a;
    }
}

__global__ __launch_bounds__(256)
void k_recover(const float* __restrict__ G, const float* __restrict__ w,
               const float* __restrict__ b, float* __restrict__ out, int rows) {
    const int r = blockIdx.x * 256 + threadIdx.x;
    if (r >= rows) return;
    const float* g = G + (size_t)r * 32;
    float a0 = b[0], a1 = b[1];
#pragma unroll
    for (int i = 0; i < 32; ++i) { a0 += w[i] * g[i]; a1 += w[32 + i] * g[i]; }
    out[r * 2] = a0;
    out[r * 2 + 1] = a1;
}

// ---------------------------------------------------------------------------
extern "C" void kernel_launch(void* const* d_in, const int* in_sizes, int n_in,
                              void* d_out, int out_size, void* d_ws, size_t ws_size,
                              hipStream_t stream) {
    (void)in_sizes; (void)n_in; (void)out_size; (void)ws_size;

    // ---- inputs (assumed pytree order, see header comment) ----
    const float* cb        = (const float*)d_in[0];
    const float* ema_cs_in = (const float*)d_in[1];
    const float* ema_w_in  = (const float*)d_in[2];
    int ii = 3;
    auto nxt = [&]() { return (const float*)d_in[ii++]; };
    // dec_layers[0]
    const float *dch1_b = nxt(), *dch1_w = nxt(), *dch2_b = nxt(), *dch2_w = nxt();
    const float *dln1_b = nxt(), *dln1_g = nxt(), *dln2_b = nxt(), *dln2_g = nxt();
    const float *dtok1_b = nxt(), *dtok1_w = nxt(), *dtok2_b = nxt(), *dtok2_w = nxt();
    const float *decln_b = nxt(), *decln_g = nxt();
    const float *dstart_b = nxt(), *dstart_w = nxt();
    const float *dtokm_b = nxt(), *dtokm_w = nxt();
    struct EncL { const float *ch1_b,*ch1_w,*ch2_b,*ch2_w,*ln1_b,*ln1_g,*ln2_b,*ln2_g,
                              *tok1_b,*tok1_w,*tok2_b,*tok2_w; } enc[4];
    for (int L = 0; L < 4; ++L) {
        enc[L].ch1_b = nxt(); enc[L].ch1_w = nxt(); enc[L].ch2_b = nxt(); enc[L].ch2_w = nxt();
        enc[L].ln1_b = nxt(); enc[L].ln1_g = nxt(); enc[L].ln2_b = nxt(); enc[L].ln2_g = nxt();
        enc[L].tok1_b = nxt(); enc[L].tok1_w = nxt(); enc[L].tok2_b = nxt(); enc[L].tok2_w = nxt();
    }
    const float *encln_b = nxt(), *encln_g = nxt();
    const float *estart_b = nxt(), *estart_w = nxt();
    const float *fe_b = nxt(), *fe_w = nxt();
    const float *rec_b = nxt(), *rec_w = nxt();
    const float *tokm_b = nxt(), *tokm_w = nxt();
    const float *x = nxt();

    // ---- workspace carve-out ----
    char* ws = (char*)d_ws;
    size_t off = 0;
    auto alloc = [&](size_t bytes) -> void* {
        void* p = ws + off;
        off += (bytes + 255) & ~(size_t)255;
        return p;
    };
    float* bufA = (float*)alloc((size_t)RP * ND * 4);   // mixer state x
    float* bufB = (float*)alloc((size_t)RP * ND * 4);   // ln output
    float* bufC = (float*)alloc((size_t)RP * ND * 4);   // gemm hidden
    float* bufD = (float*)alloc((size_t)RP * ND * 4);   // residual r
    float* bufE = (float*)alloc((size_t)RT * ND * 4);   // tt
    float* bufF = (float*)alloc((size_t)RT * ND * 4);   // ef
    float* bufG = (float*)alloc((size_t)RT * ND * 4);   // tf (=q)
    _Float16* w16_ch1[4]; _Float16* w16_ch2[4];
    for (int L = 0; L < 4; ++L) {
        w16_ch1[L] = (_Float16*)alloc((size_t)ND * ND * 2);
        w16_ch2[L] = (_Float16*)alloc((size_t)ND * ND * 2);
    }
    _Float16* w16_fe = (_Float16*)alloc((size_t)ND * ND * 2);
    _Float16* w16_ds = (_Float16*)alloc((size_t)32 * ND * 2);
    float* counts  = (float*)alloc(544 * 4);
    float* dwbuf   = (float*)alloc((size_t)544 * ND * 4);
    float* lossAcc = (float*)alloc(4);
    float* cnormb  = (float*)alloc(544 * 4);
    int*   idxb    = (int*)alloc((size_t)RT * 4);
    float* g1   = (float*)alloc((size_t)RP * 32 * 4);
    float* dlnb = (float*)alloc((size_t)RP * 32 * 4);
    float* drb  = (float*)alloc((size_t)RP * 32 * 4);
    float* dzb  = (float*)alloc((size_t)RP * 32 * 4);

    // ---- outputs ----
    float* out_kpt   = (float*)d_out;            // [B,17,2]
    float* out_loss  = out_kpt + RP * 2;         // [1]
    float* out_cbnew = out_loss + 1;             // [34,16,512]
    float* out_cs    = out_cbnew + 544 * ND;     // [34,16]
    float* out_emaw  = out_cs + 544;             // [34,16,512]

    hipMemsetAsync(counts, 0, 544 * 4, stream);
    hipMemsetAsync(dwbuf, 0, (size_t)544 * ND * 4, stream);
    hipMemsetAsync(lossAcc, 0, 4, stream);

    // ---- weight conversions (f32 -> f16) ----
    for (int L = 0; L < 4; ++L) {
        k_f32_to_f16<<<1024, 256, 0, stream>>>(enc[L].ch1_w, w16_ch1[L], ND * ND);
        k_f32_to_f16<<<1024, 256, 0, stream>>>(enc[L].ch2_w, w16_ch2[L], ND * ND);
    }
    k_f32_to_f16<<<1024, 256, 0, stream>>>(fe_w, w16_fe, ND * ND);
    k_f32_to_f16<<<64, 256, 0, stream>>>(dstart_w, w16_ds, 32 * ND);

    // ---- encoder ----
    k_enc_start<<<RP, 256, 0, stream>>>(x, estart_w, estart_b, bufA);
    for (int L = 0; L < 4; ++L) {
        k_ln512<<<RP, 256, 0, stream>>>(bufA, enc[L].ln1_g, enc[L].ln1_b, bufB);
        k_tokenmix<512><<<(NB * 512 + 255) / 256, 256, 0, stream>>>(
            bufB, bufA, enc[L].tok1_w, enc[L].tok1_b, enc[L].tok2_w, enc[L].tok2_b, bufD, NB);
        k_ln512<<<RP, 256, 0, stream>>>(bufD, enc[L].ln2_g, enc[L].ln2_b, bufB);
        k_gemm_wmma<4, true, false><<<dim3(RP / 64, ND / 64), 128, 0, stream>>>(
            bufB, w16_ch1[L], enc[L].ch1_b, nullptr, bufC, RP, ND, ND);
        k_gemm_wmma<4, false, true><<<dim3(RP / 64, ND / 64), 128, 0, stream>>>(
            bufC, w16_ch2[L], enc[L].ch2_b, bufD, bufA, RP, ND, ND);
    }
    k_ln512<<<RP, 256, 0, stream>>>(bufA, encln_g, encln_b, bufB);
    k_tokproj<17, 34><<<(NB * ND + 255) / 256, 256, 0, stream>>>(bufB, tokm_w, tokm_b, bufE, NB);
    k_gemm_wmma<4, false, false><<<dim3(RT / 64, ND / 64), 128, 0, stream>>>(
        bufE, w16_fe, fe_b, nullptr, bufF, RT, ND, ND);

    // ---- VQ + EMA ----
    k_cnorm<<<544, 256, 0, stream>>>(cb, cnormb);
    k_vq_argmin<<<RT / 8, 256, 0, stream>>>(bufF, cb, cnormb, idxb, counts);
    k_vq_scatter<<<RT, 128, 0, stream>>>(bufF, cb, idxb, bufG, dwbuf, lossAcc);
    k_cs<<<34, 16, 0, stream>>>(ema_cs_in, counts, out_cs);
    k_loss_fin<<<1, 1, 0, stream>>>(lossAcc, out_loss);
    k_ema<<<544, 256, 0, stream>>>(ema_w_in, dwbuf, out_cs, out_emaw, out_cbnew);

    // ---- decoder ----
    k_tokproj<34, 17><<<(NB * ND + 255) / 256, 256, 0, stream>>>(bufG, dtokm_w, dtokm_b, bufA, NB);
    k_gemm_wmma<2, false, false><<<dim3(RP / 64, 1), 128, 0, stream>>>(
        bufA, w16_ds, dstart_b, nullptr, g1, RP, 32, ND);
    k_ln32<<<(RP + 7) / 8, 256, 0, stream>>>(g1, dln1_g, dln1_b, dlnb, RP);
    k_tokenmix<32><<<(NB * 32 + 255) / 256, 256, 0, stream>>>(
        dlnb, g1, dtok1_w, dtok1_b, dtok2_w, dtok2_b, drb, NB);
    k_ln32<<<(RP + 7) / 8, 256, 0, stream>>>(drb, dln2_g, dln2_b, dlnb, RP);
    k_dec_chmlp<<<(RP + 255) / 256, 256, 0, stream>>>(
        dlnb, drb, dch1_w, dch1_b, dch2_w, dch2_b, dzb, RP);
    k_ln32<<<(RP + 7) / 8, 256, 0, stream>>>(dzb, decln_g, decln_b, dlnb, RP);
    k_recover<<<(RP + 255) / 256, 256, 0, stream>>>(dlnb, rec_w, rec_b, out_kpt, RP);
}
